// LSTM_28260884808399
// MI455X (gfx1250) — compile-verified, matrix operand order checked by reading, not run on hardware
//
#include <hip/hip_runtime.h>

#define N_INPUT  256
#define N_HIDDEN 512
#define DIN      768     // N_HIDDEN + N_INPUT
#define BATCH    64
#define SEQ      2048
#define NWG      32      // persistent workgroups (each owns 16 hidden cols)
#define WGSIZE   256     // 8 waves of 32

typedef __bf16 bf16;
typedef __attribute__((ext_vector_type(16))) __bf16 v16bf;
typedef __attribute__((ext_vector_type(8)))  __bf16 v8bf;
typedef __attribute__((ext_vector_type(8)))  float  v8f;

// Load one 16x32 bf16 WMMA operand fragment (this lane's 16 elements).
// p points at element (row, k0 + sel*8); second half is +16 elements.
__device__ __forceinline__ v16bf load_frag(const bf16* __restrict__ p) {
    v16bf f;
    ((v8bf*)&f)[0] = *(const v8bf*)(p);
    ((v8bf*)&f)[1] = *(const v8bf*)(p + 16);
    return f;
}

__device__ __forceinline__ v8f wmma_bf16(v16bf a, v16bf b, v8f c) {
    return __builtin_amdgcn_wmma_f32_16x16x32_bf16(
        false, a, false, b, (short)0, c, false, false);
}

__device__ __forceinline__ float sigf(float x) {
    return 1.0f / (1.0f + __expf(-x));
}

__global__ void cvt_f32_bf16(const float* __restrict__ src, bf16* __restrict__ dst, int n) {
    int stride = gridDim.x * blockDim.x;
    for (int i = blockIdx.x * blockDim.x + threadIdx.x; i < n; i += stride)
        dst[i] = (bf16)src[i];
}

__global__ void init_state(bf16* __restrict__ hbuf, unsigned* __restrict__ cnt) {
    int stride = gridDim.x * blockDim.x;
    int i = blockIdx.x * blockDim.x + threadIdx.x;
    if (i == 0) *cnt = 0u;
    for (int j = i; j < 3 * BATCH * N_HIDDEN; j += stride)
        hbuf[j] = (bf16)0.0f;
}

__global__ __launch_bounds__(WGSIZE) void lstm_persistent(
    const bf16* __restrict__ xbf,     // [BATCH][SEQ][N_INPUT]
    const bf16* __restrict__ Wg,      // [4*N_HIDDEN][DIN] (f,i,g,o stacked)
    const bf16* __restrict__ Wout,    // [N_HIDDEN][N_HIDDEN]
    bf16* __restrict__ hbuf,          // [3][BATCH][N_HIDDEN] triple buffer
    unsigned* __restrict__ barrier_cnt,
    const float* __restrict__ b_f, const float* __restrict__ b_i,
    const float* __restrict__ b_g, const float* __restrict__ b_o,
    const float* __restrict__ b_out,
    float* __restrict__ out)          // [BATCH][SEQ][N_HIDDEN]
{
    __shared__ float lgates[4][BATCH][16];  // gate pre-activations for this WG's slice
    __shared__ float c_lds[BATCH][16];      // persistent cell state (f32)

    const int w    = blockIdx.x;       // hidden slice: cols [w*16, w*16+16)
    const int tid  = threadIdx.x;
    const int wave = tid >> 5;         // 0..7 (wave32)
    const int lane = tid & 31;
    const int sel  = lane >> 4;        // which K half-group this lane holds
    const int r    = lane & 15;        // row-within-tile (A: M, B: N, C: col)

    for (int e = tid; e < BATCH * 16; e += WGSIZE)
        c_lds[e >> 4][e & 15] = 0.0f;
    __syncthreads();

    const int gate = wave >> 1;        // 0..3 -> f,i,g,o
    const int mt0  = (wave & 1) * 2;   // M-tiles {mt0, mt0+1}
    const int m0   = mt0 * 16 + r;
    const int m1   = m0 + 16;

    const bf16* Bg_row = Wg   + (size_t)(gate * N_HIDDEN + w * 16 + r) * DIN;
    const bf16* BO_row = Wout + (size_t)(w * 16 + r) * N_HIDDEN;

    const v8f zf = {};
    v8f acc0 = zf, acc1 = zf;

    // x-contribution of the gate GEMM for step tt (k = 512..767), accumulated
    // into acc0/acc1. Independent of h -> runs inside the barrier spin window.
    auto gate_xpart = [&](int tt) {
        acc0 = zf; acc1 = zf;
        for (int ks = 0; ks < (DIN - N_HIDDEN) / 32; ++ks) {
            const int kx = ks * 32 + sel * 8;
            v16bf b  = load_frag(Bg_row + N_HIDDEN + kx);
            v16bf a0 = load_frag(xbf + ((size_t)m0 * SEQ + tt) * N_INPUT + kx);
            v16bf a1 = load_frag(xbf + ((size_t)m1 * SEQ + tt) * N_INPUT + kx);
            acc0 = wmma_bf16(a0, b, acc0);
            acc1 = wmma_bf16(a1, b, acc1);
        }
    };

    // out[:, tt, w*16..+16) = relu(h_tt . Wout^T + b_out); waves 0..3 (uniform).
    auto do_out = [&](int tt, const bf16* __restrict__ hsrc) {
        if (wave < 4) {
            v8f acc = zf;
            const int ma = wave * 16 + r;
            for (int ks = 0; ks < N_HIDDEN / 32; ++ks) {
                const int k0 = ks * 32 + sel * 8;
                v16bf b = load_frag(BO_row + k0);
                v16bf a = load_frag(hsrc + (size_t)ma * N_HIDDEN + k0);
                acc = wmma_bf16(a, b, acc);
            }
#pragma unroll
            for (int q = 0; q < 8; ++q) {
                const int m   = wave * 16 + sel * 8 + q;
                const int col = w * 16 + r;
                float v = acc[q] + b_out[col];
                out[((size_t)m * SEQ + tt) * N_HIDDEN + col] = v > 0.0f ? v : 0.0f;
            }
        }
    };

    gate_xpart(0);   // prologue: x-part for t = 0

    for (int t = 0; t < SEQ; ++t) {
        const bf16* hprev = hbuf + (size_t)((t + 2) % 3) * (BATCH * N_HIDDEN); // h_{t-1}
        bf16*       hcur  = hbuf + (size_t)(t % 3) * (BATCH * N_HIDDEN);       // h_t

        // ---- gate GEMM, h-part (k = 0..511); acc already holds x-part ----
        for (int ks = 0; ks < N_HIDDEN / 32; ++ks) {
            const int k0 = ks * 32 + sel * 8;
            v16bf b  = load_frag(Bg_row + k0);
            v16bf a0 = load_frag(hprev + (size_t)m0 * N_HIDDEN + k0);
            v16bf a1 = load_frag(hprev + (size_t)m1 * N_HIDDEN + k0);
            acc0 = wmma_bf16(a0, b, acc0);
            acc1 = wmma_bf16(a1, b, acc1);
        }
        // C/D layout: lane holds (M = q + 8*sel, N = r) for q = 0..7
#pragma unroll
        for (int q = 0; q < 8; ++q) {
            lgates[gate][mt0 * 16 + sel * 8 + q][r]       = acc0[q];
            lgates[gate][(mt0 + 1) * 16 + sel * 8 + q][r] = acc1[q];
        }
        __syncthreads();

        // ---- fused elementwise cell update for this WG's 64x16 slice ----
        for (int e = tid; e < BATCH * 16; e += WGSIZE) {
            const int m = e >> 4, j = e & 15;
            const int col = w * 16 + j;
            const float fg = sigf(lgates[0][m][j] + b_f[col]);
            const float ig = sigf(lgates[1][m][j] + b_i[col]);
            const float gg = sigf(lgates[2][m][j] + b_g[col]);  // sigmoid per reference
            const float og = sigf(lgates[3][m][j] + b_o[col]);
            const float c  = c_lds[m][j] * fg + ig * gg;
            c_lds[m][j] = c;
            const float th = 1.0f - 2.0f / (__expf(2.0f * c) + 1.0f);  // tanh(c)
            hcur[(size_t)m * N_HIDDEN + col] = (bf16)(og * th);
        }

        // ---- barrier arrive (release h_t) ----
        __threadfence();        // all threads release their h stores
        __syncthreads();
        if (tid == 0)
            __hip_atomic_fetch_add(barrier_cnt, 1u, __ATOMIC_RELEASE,
                                   __HIP_MEMORY_SCOPE_AGENT);

        // ---- overlap window: work independent of the barrier ----
        if (t + 1 < SEQ) gate_xpart(t + 1);  // x-part for next step
        else             { acc0 = zf; acc1 = zf; }
        if (t > 0) do_out(t - 1, hprev);     // h_{t-1} fully visible; buffer safe
                                             // until elementwise(t+2) > barrier(t+1)

        // ---- barrier wait (acquire all WGs' h_t) ----
        if (tid == 0) {
            const unsigned target = (unsigned)(t + 1) * NWG;
            while (__hip_atomic_load(barrier_cnt, __ATOMIC_ACQUIRE,
                                     __HIP_MEMORY_SCOPE_AGENT) < target)
                __builtin_amdgcn_s_sleep(1);
        }
        __syncthreads();
        __threadfence();
    }

    // epilogue: output for the final step (h_{SEQ-1} visible after last barrier)
    do_out(SEQ - 1, hbuf + (size_t)((SEQ - 1) % 3) * (BATCH * N_HIDDEN));
}

extern "C" void kernel_launch(void* const* d_in, const int* in_sizes, int n_in,
                              void* d_out, int out_size, void* d_ws, size_t ws_size,
                              hipStream_t stream) {
    (void)in_sizes; (void)n_in; (void)out_size; (void)ws_size;

    const float* input = (const float*)d_in[0];
    const float* W_f   = (const float*)d_in[1];
    const float* b_f   = (const float*)d_in[2];
    const float* W_i   = (const float*)d_in[3];
    const float* b_i   = (const float*)d_in[4];
    const float* W_g   = (const float*)d_in[5];
    const float* b_g   = (const float*)d_in[6];
    const float* W_o   = (const float*)d_in[7];
    const float* b_o   = (const float*)d_in[8];
    const float* W_out = (const float*)d_in[9];
    const float* b_out = (const float*)d_in[10];

    char* ws = (char*)d_ws;
    size_t off = 0;
    bf16* xbf    = (bf16*)(ws + off); off += (size_t)BATCH * SEQ * N_INPUT * 2;  // 64 MB
    bf16* Wgbf   = (bf16*)(ws + off); off += (size_t)4 * N_HIDDEN * DIN * 2;     // 3 MB
    bf16* Woutbf = (bf16*)(ws + off); off += (size_t)N_HIDDEN * N_HIDDEN * 2;    // 0.5 MB
    bf16* hbuf   = (bf16*)(ws + off); off += (size_t)3 * BATCH * N_HIDDEN * 2;   // 192 KB
    unsigned* barrier_cnt = (unsigned*)(ws + off);

    // bf16 conversion (idempotent, re-run every call)
    cvt_f32_bf16<<<2048, 256, 0, stream>>>(input, xbf, BATCH * SEQ * N_INPUT);
    cvt_f32_bf16<<<512, 256, 0, stream>>>(W_f, Wgbf + (size_t)0 * N_HIDDEN * DIN, N_HIDDEN * DIN);
    cvt_f32_bf16<<<512, 256, 0, stream>>>(W_i, Wgbf + (size_t)1 * N_HIDDEN * DIN, N_HIDDEN * DIN);
    cvt_f32_bf16<<<512, 256, 0, stream>>>(W_g, Wgbf + (size_t)2 * N_HIDDEN * DIN, N_HIDDEN * DIN);
    cvt_f32_bf16<<<512, 256, 0, stream>>>(W_o, Wgbf + (size_t)3 * N_HIDDEN * DIN, N_HIDDEN * DIN);
    cvt_f32_bf16<<<256, 256, 0, stream>>>(W_out, Woutbf, N_HIDDEN * N_HIDDEN);
    init_state<<<64, 256, 0, stream>>>(hbuf, barrier_cnt);

    lstm_persistent<<<NWG, WGSIZE, 0, stream>>>(
        xbf, Wgbf, Woutbf, hbuf, barrier_cnt,
        b_f, b_i, b_g, b_o, b_out, (float*)d_out);
}
